// MemoryModule_40192303956211
// MI455X (gfx1250) — matrix operand order repaired and checked by reading, not compiled, not run on hardware
//
#include <hip/hip_runtime.h>
#include <math.h>

// ---------------------------------------------------------------------------
// MemoryModule for MI455X (gfx1250, wave32, WMMA bf16 path)
// B=4, T=4096 -> Ntok=16384 ; D=1024 ; S=2048 ; K=8 ; H=512
// ---------------------------------------------------------------------------

typedef __bf16 bf16_t;
typedef __attribute__((ext_vector_type(16))) __bf16 v16bf;
typedef __attribute__((ext_vector_type(8)))  __bf16 v8bf;
typedef __attribute__((ext_vector_type(8)))  float  v8f;

#define D_DIM 1024
#define S_DIM 2048
#define K_TOP 8
#define H_DIM 512

// ------------------------- WMMA fragment helpers ---------------------------
// A fragment: 16x32 bf16, lane 0-15 -> rows, K contiguous (matches B):
//   lanes 0-15 : K 0..15
//   lanes 16-31: K 16..31
__device__ __forceinline__ v16bf load_a_frag(const bf16_t* __restrict__ A, int lda,
                                             int m0, int k0, int lane) {
  const int half = lane >> 4;
  const int r = m0 + (lane & 15);
  const bf16_t* p = A + (size_t)r * lda + k0 + half * 16;
  v8bf lo = *(const v8bf*)(p);        // K = k0 + half*16 + [0..7]
  v8bf hi = *(const v8bf*)(p + 8);    // K = k0 + half*16 + 8 + [0..7]
  v16bf f;
#pragma unroll
  for (int i = 0; i < 8; ++i) { f[i] = lo[i]; f[i + 8] = hi[i]; }
  return f;
}

// B fragment: 32x16 bf16 (B[k][n] = W[n][k], W row-major [Nd, Kd]):
//   lane 0-15 : col n = n0+lane,    K = k0 + [0..15]
//   lane 16-31: col n = n0+lane-16, K = k0 + 16 + [0..15]
__device__ __forceinline__ v16bf load_b_frag(const bf16_t* __restrict__ W, int ldw,
                                             int n0, int k0, int lane) {
  const bf16_t* p = W + (size_t)(n0 + (lane & 15)) * ldw + k0 + (lane >> 4) * 16;
  return *(const v16bf*)(p);
}

__device__ __forceinline__ v8f wmma_bf16(v16bf a, v16bf b, v8f c) {
  return __builtin_amdgcn_wmma_f32_16x16x32_bf16(false, a, false, b,
                                                 (short)0, c, false, false);
}

// C/D layout: lane L -> col n0+(L&15); VGPR r -> row m0 + r + 8*(L>>4)
__device__ __forceinline__ void store_c(float* __restrict__ C, int ldc,
                                        int m0, int n0, int lane, v8f acc) {
  float* cp = C + (size_t)(m0 + 8 * (lane >> 4)) * ldc + n0 + (lane & 15);
#pragma unroll
  for (int r = 0; r < 8; ++r) cp[(size_t)r * ldc] = acc[r];
}

// ------------------------- generic WMMA GEMM: C = A @ W^T ------------------
// Block tile 128(M) x 64(N); 8 waves arranged 4x2; wave tile 32x32 (2x2 WMMA).
// All dims (M=16384, Nd in {512,1024}, Kd in {1024,2048}) divide the tiling.
__global__ void __launch_bounds__(256)
wmma_gemm_nt(const bf16_t* __restrict__ A, int lda,
             const bf16_t* __restrict__ W, int ldw,
             float* __restrict__ C, int ldc, int Kd) {
  const int lane = threadIdx.x & 31;
  const int wave = threadIdx.x >> 5;
  const int wm = wave & 3;
  const int wn = wave >> 2;
  const int m0 = blockIdx.x * 128 + wm * 32;
  const int n0 = blockIdx.y * 64 + wn * 32;

  v8f acc00 = {0.f,0.f,0.f,0.f,0.f,0.f,0.f,0.f};
  v8f acc01 = acc00, acc10 = acc00, acc11 = acc00;

  for (int k0 = 0; k0 < Kd; k0 += 32) {
    v16bf a0 = load_a_frag(A, lda, m0,      k0, lane);
    v16bf a1 = load_a_frag(A, lda, m0 + 16, k0, lane);
    v16bf b0 = load_b_frag(W, ldw, n0,      k0, lane);
    v16bf b1 = load_b_frag(W, ldw, n0 + 16, k0, lane);
    acc00 = wmma_bf16(a0, b0, acc00);
    acc01 = wmma_bf16(a0, b1, acc01);
    acc10 = wmma_bf16(a1, b0, acc10);
    acc11 = wmma_bf16(a1, b1, acc11);
  }
  store_c(C, ldc, m0,      n0,      lane, acc00);
  store_c(C, ldc, m0,      n0 + 16, lane, acc01);
  store_c(C, ldc, m0 + 16, n0,      lane, acc10);
  store_c(C, ldc, m0 + 16, n0 + 16, lane, acc11);
}

// ------------------------- fp32 -> bf16 (strided 2D) -----------------------
__global__ void cvt_f32_bf16_2d(const float* __restrict__ src, int src_ld,
                                bf16_t* __restrict__ dst, int dst_ld,
                                int rows, int cols) {
  const long long total = (long long)rows * cols;
  for (long long i = (long long)blockIdx.x * blockDim.x + threadIdx.x; i < total;
       i += (long long)gridDim.x * blockDim.x) {
    const long long r = i / cols;
    const long long c = i % cols;
    dst[r * dst_ld + c] = (bf16_t)src[r * src_ld + c];
  }
}

// ------------------------- row L2 normalize -> bf16 ------------------------
__global__ void __launch_bounds__(256)
l2norm_rows_bf16(const float* __restrict__ in, bf16_t* __restrict__ out, int cols) {
  const int row = blockIdx.x;
  const float* p = in + (size_t)row * cols;
  float ss = 0.f;
  for (int j = threadIdx.x; j < cols; j += 256) { const float v = p[j]; ss += v * v; }
#pragma unroll
  for (int off = 16; off > 0; off >>= 1) ss += __shfl_xor(ss, off, 32);
  __shared__ float red[8];
  __shared__ float invn;
  const int lane = threadIdx.x & 31, wave = threadIdx.x >> 5;
  if (lane == 0) red[wave] = ss;
  __syncthreads();
  if (threadIdx.x == 0) {
    float s = 0.f;
    for (int w = 0; w < 8; ++w) s += red[w];
    invn = 1.0f / fmaxf(sqrtf(s), 1e-12f);
  }
  __syncthreads();
  const float inv = invn;
  bf16_t* o = out + (size_t)row * cols;
  for (int j = threadIdx.x; j < 256 * ((cols + 255) / 256); j += 256)
    if (j < cols) o[j] = (bf16_t)(p[j] * inv);
}

// ---------------- sims (WMMA) + top-8 + softmax + value gather -------------
// Grid: Ntok/16 blocks of 256 threads (8 waves). Dynamic LDS:
//   sims[16][S_DIM] fp32 = 128 KB (fits 320 KB WGP LDS), + topk idx/weights.
__global__ void __launch_bounds__(256)
sims_topk_retrieve(const bf16_t* __restrict__ qn,   // [Ntok, D] normalized bf16
                   const bf16_t* __restrict__ kn,   // [S, D]   normalized bf16
                   const float*  __restrict__ V,    // [S, D]   fp32 values
                   bf16_t* __restrict__ retb) {     // [Ntok, D] bf16 out
  extern __shared__ char smem_raw[];
  float* sims = (float*)smem_raw;                               // [16][S_DIM]
  int*   tidx = (int*)(smem_raw + (size_t)16 * S_DIM * 4);      // [16][8]
  float* tw   = (float*)(smem_raw + (size_t)16 * S_DIM * 4 + 16 * 8 * 4); // [16][8]

  const int lane = threadIdx.x & 31;
  const int wave = threadIdx.x >> 5;
  const int m0 = blockIdx.x * 16;

  // Phase 1: sims[16][2048] = Qn_tile @ Kn^T via WMMA, staged to LDS only.
  for (int st = wave; st < S_DIM / 16; st += 8) {
    const int n0 = st * 16;
    v8f acc = {0.f,0.f,0.f,0.f,0.f,0.f,0.f,0.f};
    for (int k0 = 0; k0 < D_DIM; k0 += 32) {
      v16bf a = load_a_frag(qn, D_DIM, m0, k0, lane);
      v16bf b = load_b_frag(kn, D_DIM, n0, k0, lane);
      acc = wmma_bf16(a, b, acc);
    }
    const int col = n0 + (lane & 15);
    const int rb = 8 * (lane >> 4);
#pragma unroll
    for (int r = 0; r < 8; ++r) sims[(rb + r) * S_DIM + col] = acc[r];
  }
  __syncthreads();

  // Phase 2: per wave, 2 rows: iterative wave-wide argmax x8, then softmax.
  for (int rr = 0; rr < 2; ++rr) {
    const int row = wave * 2 + rr;
    float* srow = sims + (size_t)row * S_DIM;
    for (int k = 0; k < K_TOP; ++k) {
      float best = -1e30f; int bi = S_DIM;
      for (int j = lane; j < S_DIM; j += 32) {
        const float v = srow[j];
        if (v > best || (v == best && j < bi)) { best = v; bi = j; }
      }
#pragma unroll
      for (int off = 16; off > 0; off >>= 1) {
        const float ov = __shfl_xor(best, off, 32);
        const int   oi = __shfl_xor(bi, off, 32);
        if (ov > best || (ov == best && oi < bi)) { best = ov; bi = oi; }
      }
      if (lane == 0) { tidx[row * 8 + k] = bi; tw[row * 8 + k] = best; srow[bi] = -1e30f; }
      // lane0's DS store is in-order with this wave's subsequent DS loads.
    }
    if (lane == 0) {
      float mx = tw[row * 8];
      for (int k = 1; k < K_TOP; ++k) mx = fmaxf(mx, tw[row * 8 + k]);
      float e[K_TOP]; float s = 0.f;
      for (int k = 0; k < K_TOP; ++k) { e[k] = __expf(tw[row * 8 + k] - mx); s += e[k]; }
      const float inv = 1.0f / s;
      for (int k = 0; k < K_TOP; ++k) tw[row * 8 + k] = e[k] * inv;
    }
  }

  // Phase 3: gather + weighted sum of value rows (float4 vectorized).
  for (int rr = 0; rr < 2; ++rr) {
    const int row = wave * 2 + rr;
    float4 acc4[8];
#pragma unroll
    for (int j = 0; j < 8; ++j) acc4[j] = make_float4(0.f, 0.f, 0.f, 0.f);
#pragma unroll
    for (int k = 0; k < K_TOP; ++k) {
      const int s = tidx[row * 8 + k];
      const float w = tw[row * 8 + k];
      const float4* vp = (const float4*)(V + (size_t)s * D_DIM);
#pragma unroll
      for (int j = 0; j < 8; ++j) {
        const float4 v = vp[lane + 32 * j];
        acc4[j].x += w * v.x; acc4[j].y += w * v.y;
        acc4[j].z += w * v.z; acc4[j].w += w * v.w;
      }
    }
    bf16_t* op = retb + (size_t)(m0 + row) * D_DIM;
#pragma unroll
    for (int j = 0; j < 8; ++j) {
      const int d = (lane + 32 * j) * 4;
      op[d + 0] = (bf16_t)acc4[j].x; op[d + 1] = (bf16_t)acc4[j].y;
      op[d + 2] = (bf16_t)acc4[j].z; op[d + 3] = (bf16_t)acc4[j].w;
    }
  }
}

// ---------------- gate: bias + GELU(exact) + dot(gW2) + sigmoid + residual -
__global__ void __launch_bounds__(128)
gate_out(const float* __restrict__ x, const float* __restrict__ hpre,
         const float* __restrict__ gb1, const float* __restrict__ gW2,
         const float* __restrict__ gb2, const float* __restrict__ retWo,
         float* __restrict__ out) {
  const int t = blockIdx.x;
  const float* hp = hpre + (size_t)t * H_DIM;
  float acc = 0.f;
  for (int j = threadIdx.x; j < H_DIM; j += 128) {
    const float pre = hp[j] + gb1[j];
    const float g = 0.5f * pre * (1.0f + erff(pre * 0.70710678118654752f));
    acc += g * gW2[j];
  }
#pragma unroll
  for (int off = 16; off > 0; off >>= 1) acc += __shfl_xor(acc, off, 32);
  __shared__ float red[4];
  __shared__ float gsh;
  const int lane = threadIdx.x & 31, wave = threadIdx.x >> 5;
  if (lane == 0) red[wave] = acc;
  __syncthreads();
  if (threadIdx.x == 0) {
    const float s = red[0] + red[1] + red[2] + red[3] + gb2[0];
    gsh = 1.0f / (1.0f + __expf(-s));
  }
  __syncthreads();
  const float gate = gsh;
  const float* xr = x + (size_t)t * D_DIM;
  const float* rw = retWo + (size_t)t * D_DIM;
  float* orow = out + (size_t)t * D_DIM;
  for (int j = threadIdx.x; j < D_DIM; j += 128) orow[j] = xr[j] + gate * rw[j];
}

// ---------------------------------------------------------------------------
extern "C" void kernel_launch(void* const* d_in, const int* in_sizes, int n_in,
                              void* d_out, int out_size, void* d_ws, size_t ws_size,
                              hipStream_t stream) {
  const float* x   = (const float*)d_in[0];
  const float* mk  = (const float*)d_in[1];
  const float* mv  = (const float*)d_in[2];
  const float* Wq  = (const float*)d_in[3];
  const float* Wo  = (const float*)d_in[4];
  const float* gW1 = (const float*)d_in[5];
  const float* gb1 = (const float*)d_in[6];
  const float* gW2 = (const float*)d_in[7];
  const float* gb2 = (const float*)d_in[8];
  float* out = (float*)d_out;
  const int Nt = in_sizes[0] / D_DIM;   // 16384

  char* ws = (char*)d_ws;
  size_t off = 0;
  auto alloc = [&](size_t bytes) -> char* {
    char* p = ws + off;
    off = (off + bytes + 255) & ~(size_t)255;
    return p;
  };
  bf16_t* gatein = (bf16_t*)alloc((size_t)Nt * 2 * D_DIM * 2); // [N,2D] bf16 concat
  bf16_t* wqb    = (bf16_t*)alloc((size_t)D_DIM * D_DIM * 2);
  bf16_t* wob    = (bf16_t*)alloc((size_t)D_DIM * D_DIM * 2);
  bf16_t* gw1b   = (bf16_t*)alloc((size_t)H_DIM * 2 * D_DIM * 2);
  bf16_t* knb    = (bf16_t*)alloc((size_t)S_DIM * D_DIM * 2);
  float*  qf32   = (float*)alloc((size_t)Nt * D_DIM * 4);      // queries, reused as retWo
  bf16_t* qnb    = (bf16_t*)alloc((size_t)Nt * D_DIM * 2);
  bf16_t* retb   = (bf16_t*)alloc((size_t)Nt * D_DIM * 2);
  float*  hpre   = (float*)alloc((size_t)Nt * H_DIM * 4);
  (void)ws_size; (void)n_in; (void)out_size;

  // bf16 conversions (x lands in the first half of the concat buffer)
  cvt_f32_bf16_2d<<<2048, 256, 0, stream>>>(x,  D_DIM, gatein, 2 * D_DIM, Nt, D_DIM);
  cvt_f32_bf16_2d<<<512,  256, 0, stream>>>(Wq, D_DIM, wqb,    D_DIM, D_DIM, D_DIM);
  cvt_f32_bf16_2d<<<512,  256, 0, stream>>>(Wo, D_DIM, wob,    D_DIM, D_DIM, D_DIM);
  cvt_f32_bf16_2d<<<512,  256, 0, stream>>>(gW1, 2 * D_DIM, gw1b, 2 * D_DIM, H_DIM, 2 * D_DIM);

  // normalized keys -> bf16
  l2norm_rows_bf16<<<S_DIM, 256, 0, stream>>>(mk, knb, D_DIM);

  // queries = x @ Wq^T  (WMMA)
  wmma_gemm_nt<<<dim3(Nt / 128, D_DIM / 64), 256, 0, stream>>>(
      gatein, 2 * D_DIM, wqb, D_DIM, qf32, D_DIM, D_DIM);

  // normalized queries -> bf16
  l2norm_rows_bf16<<<Nt, 256, 0, stream>>>(qf32, qnb, D_DIM);

  // sims (WMMA, LDS-resident) + top-8 + softmax + value gather
  const size_t smem = (size_t)16 * S_DIM * 4 + (size_t)16 * 8 * 4 * 2;
  sims_topk_retrieve<<<Nt / 16, 256, smem, stream>>>(qnb, knb, mv, retb);

  // retWo = retrieved @ Wo^T  (WMMA) -> reuse qf32 buffer
  wmma_gemm_nt<<<dim3(Nt / 128, D_DIM / 64), 256, 0, stream>>>(
      retb, D_DIM, wob, D_DIM, qf32, D_DIM, D_DIM);

  // second half of concat buffer = bf16(retWo)
  cvt_f32_bf16_2d<<<2048, 256, 0, stream>>>(qf32, D_DIM, gatein + D_DIM, 2 * D_DIM, Nt, D_DIM);

  // hpre = [x, retWo] @ gW1^T  (WMMA)
  wmma_gemm_nt<<<dim3(Nt / 128, H_DIM / 64), 256, 0, stream>>>(
      gatein, 2 * D_DIM, gw1b, 2 * D_DIM, hpre, H_DIM, 2 * D_DIM);

  // gate + residual
  gate_out<<<Nt, 128, 0, stream>>>(x, hpre, gb1, gW2, gb2, qf32, out);
}
